// PlasticBrain_35940286333496
// MI455X (gfx1250) — matrix operand order repaired and checked by reading, not compiled, not run on hardware
//
#include <hip/hip_runtime.h>
#include <hip/hip_bf16.h>

typedef __attribute__((ext_vector_type(16))) __bf16 v16bf;
typedef __attribute__((ext_vector_type(8)))  __bf16 v8bf;
typedef __attribute__((ext_vector_type(8)))  float  v8f;

#define D_DIM       512
#define ROWS        32        // batch rows per block (2 row tiles)
#define CHUNK       1024      // key columns per block
#define COLS_IT     64        // key columns staged per iteration
#define NIT         (CHUNK / COLS_IT)     // 16
#define BSTRIDE     520       // padded bf16 column stride in LDS (bank rotation)
#define BUF_ELEMS   (COLS_IT * BSTRIDE)   // 33280 bf16 = 66560 B per buffer
#define SC_BYTES    (ROWS * CHUNK * 4)    // 131072 B
#define LDS_BYTES   (SC_BYTES + 2 * BUF_ELEMS * 2)   // 264192 B (< 320 KB)
#define K_NB        50
#define NUM_CLASSES 1000
#define NEG_FILL    (-1.0e9f)
#define NEG_INF     (-3.0e38f)
#define CAND_MAX    6400      // (M/CHUNK)*K_NB = 128*50

// ---------------------------------------------------------------------------
// Kernel 1: L2-normalize each row (fp32) and emit bf16. One wave per row.
// ---------------------------------------------------------------------------
__global__ __launch_bounds__(256) void rownorm_bf16(const float* __restrict__ in,
                                                    __bf16* __restrict__ out,
                                                    int nrows) {
  const int wave = threadIdx.x >> 5;
  const int lane = threadIdx.x & 31;
  const int row  = blockIdx.x * 8 + wave;
  if (row >= nrows) return;

  const float* r = in + (size_t)row * D_DIM;
  float ss = 0.0f;
#pragma unroll
  for (int j = lane; j < D_DIM; j += 32) { float v = r[j]; ss += v * v; }
#pragma unroll
  for (int off = 16; off; off >>= 1) ss += __shfl_xor(ss, off, 32);

  const float scale = 1.0f / fmaxf(sqrtf(ss), 1e-12f);  // matches F.normalize eps
  __bf16* o = out + (size_t)row * D_DIM;
#pragma unroll
  for (int j = lane; j < D_DIM; j += 32) o[j] = (__bf16)(r[j] * scale);
}

// ---------------------------------------------------------------------------
// Async stage: copy 64 key rows (512 bf16 each) into LDS with padded stride.
// 256 threads x 16 quanta of 16B each, via the CDNA5 async-to-LDS engine.
// ---------------------------------------------------------------------------
__device__ __forceinline__ void stage_tile(const __bf16* __restrict__ kn,
                                           size_t gElemBase,   // element offset of col0
                                           char* ldsBase,      // generic ptr into LDS buf
                                           int tid) {
#pragma unroll 1
  for (int u = tid; u < COLS_IT * 64; u += 256) {     // 64 cols * 64 16B-quanta
    const int c = u >> 6;
    const int q = u & 63;
    unsigned ldsoff = (unsigned)(uintptr_t)(ldsBase + ((size_t)c * BSTRIDE + q * 8) * 2);
    const __bf16* g = kn + gElemBase + (size_t)c * D_DIM + q * 8;
    asm volatile("global_load_async_to_lds_b128 %0, %1, off"
                 :: "v"(ldsoff), "v"((unsigned long long)(uintptr_t)g)
                 : "memory");
  }
}

// ---------------------------------------------------------------------------
// Kernel 2: WMMA GEMM (32 rows x CHUNK cols, K=512) + per-chunk top-50.
// 8 waves: wave w -> row tile (w>>2), 16-col tile (w&3) of a 64-col group.
// Key tiles double-buffered in LDS via global_load_async_to_lds_b128;
// compute reads B fragments from LDS while the next tile streams in.
// ---------------------------------------------------------------------------
__global__ __launch_bounds__(256) void gemm_topk(const __bf16* __restrict__ xn,
                                                 const __bf16* __restrict__ kn,
                                                 const int*    __restrict__ values,
                                                 float* __restrict__ candScore,
                                                 int*   __restrict__ candIdx,
                                                 int nChunks) {
  extern __shared__ char smem[];
  float*  sc  = (float*)smem;                       // [ROWS][CHUNK] scores
  __bf16* bst = (__bf16*)(smem + SC_BYTES);         // [2][COLS_IT][BSTRIDE]

  const int rowBase  = blockIdx.x * ROWS;
  const int chunk    = blockIdx.y;
  const int colBase0 = chunk * CHUNK;
  const int tid    = threadIdx.x;
  const int wave   = tid >> 5;
  const int lane   = tid & 31;
  const int laneM  = lane & 15;
  const int laneHi = lane >> 4;
  const int rt = wave >> 2;                         // row tile 0..1
  const int ct = wave & 3;                          // col tile 0..3

  // A fragment base: lane half 0 -> K {0..7,16..23}, half 1 -> K {8..15,24..31}
  const __bf16* arow = xn + (size_t)(rowBase + rt * 16 + laneM) * D_DIM + laneHi * 8;

  // Prologue: stage tile 0, wait, publish.
  stage_tile(kn, (size_t)colBase0 * D_DIM, (char*)bst, tid);
  asm volatile("s_wait_asynccnt 0x0" ::: "memory");
  __syncthreads();

  int buf = 0;
#pragma unroll 1
  for (int it = 0; it < NIT; ++it) {
    if (it + 1 < NIT)                               // overlap: stream next 64 cols
      stage_tile(kn, (size_t)(colBase0 + (it + 1) * COLS_IT) * D_DIM,
                 (char*)(bst + (size_t)(buf ^ 1) * BUF_ELEMS), tid);

    // B fragment from LDS: lane col = ct*16+laneM, K block split by lane half.
    const __bf16* bcol = bst + (size_t)buf * BUF_ELEMS
                       + (size_t)(ct * 16 + laneM) * BSTRIDE + laneHi * 8;
    v8f acc = {};
#pragma unroll
    for (int k = 0; k < D_DIM; k += 32) {
      v8bf blo = *(const v8bf*)(bcol + k);
      v8bf bhi = *(const v8bf*)(bcol + k + 16);
      v16bf b  = __builtin_shufflevector(blo, bhi, 0,1,2,3,4,5,6,7,
                                                   8,9,10,11,12,13,14,15);
      v8bf a0 = *(const v8bf*)(arow + k);
      v8bf a1 = *(const v8bf*)(arow + k + 16);
      v16bf a = __builtin_shufflevector(a0, a1, 0,1,2,3,4,5,6,7,
                                                8,9,10,11,12,13,14,15);
      acc = __builtin_amdgcn_wmma_f32_16x16x32_bf16(
                false, a, false, b, (short)0, acc, false, false);
    }

    const int colLocal = it * COLS_IT + ct * 16 + laneM;
    const bool empty = (values[colBase0 + colLocal] == -1);
#pragma unroll
    for (int i = 0; i < 8; ++i) {                   // D: VGPR i -> row i (+8 hi half)
      float s = empty ? NEG_FILL : acc[i];
      sc[(rt * 16 + i + 8 * laneHi) * CHUNK + colLocal] = s;
    }

    asm volatile("s_wait_asynccnt 0x0" ::: "memory");  // next buffer landed
    __syncthreads();
    buf ^= 1;
  }

  // Per-row top-50 over CHUNK scores in LDS. 8 waves x 4 rows each.
  for (int rr = 0; rr < 4; ++rr) {
    const int rloc = wave * 4 + rr;
    float* row = sc + (size_t)rloc * CHUNK;
    for (int k = 0; k < K_NB; ++k) {
      float best = NEG_INF; int bi = lane;
      for (int j = lane; j < CHUNK; j += 32) {      // conflict-free stride
        float v = row[j];
        if (v > best) { best = v; bi = j; }
      }
#pragma unroll
      for (int off = 16; off; off >>= 1) {          // wave32 butterfly argmax
        float ob = __shfl_xor(best, off, 32);
        int   oi = __shfl_xor(bi,   off, 32);
        if (ob > best || (ob == best && oi < bi)) { best = ob; bi = oi; }
      }
      if (lane == 0) {
        size_t o = (((size_t)(rowBase + rloc)) * nChunks + chunk) * K_NB + k;
        candScore[o] = best;
        candIdx[o]   = colBase0 + bi;
        row[bi] = NEG_INF;                          // same-wave LDS ops are in order
      }
    }
  }
}

// ---------------------------------------------------------------------------
// Kernel 3: per batch row, merge nChunks*50 candidates -> exact top-50,
// then soft-vote into the class histogram. One block per row.
// ---------------------------------------------------------------------------
__global__ __launch_bounds__(256) void merge_vote(const float* __restrict__ candScore,
                                                  const int*   __restrict__ candIdx,
                                                  const int*   __restrict__ values,
                                                  float* __restrict__ logits,
                                                  int nChunks) {
  __shared__ float s [CAND_MAX];
  __shared__ int   si[CAND_MAX];
  __shared__ float hist[NUM_CLASSES];

  const int row = blockIdx.x;
  const int n   = nChunks * K_NB;                   // 6400
  for (int j = threadIdx.x; j < n; j += 256) {
    s [j] = candScore[(size_t)row * n + j];
    si[j] = candIdx [(size_t)row * n + j];
  }
  for (int j = threadIdx.x; j < NUM_CLASSES; j += 256) hist[j] = 0.0f;
  __syncthreads();

  if (threadIdx.x < 32) {                           // wave 0 does the selection
    const int lane = threadIdx.x;
    for (int k = 0; k < K_NB; ++k) {
      float best = NEG_INF; int bi = lane;
      for (int j = lane; j < n; j += 32) {
        float v = s[j];
        if (v > best) { best = v; bi = j; }
      }
#pragma unroll
      for (int off = 16; off; off >>= 1) {
        float ob = __shfl_xor(best, off, 32);
        int   oi = __shfl_xor(bi,   off, 32);
        if (ob > best || (ob == best && oi < bi)) { best = ob; bi = oi; }
      }
      if (lane == 0) {
        int lbl = values[si[bi]];
        if (lbl != -1) hist[lbl] += best;           // reference skips invalid labels
        s[bi] = NEG_INF;
      }
    }
  }
  __syncthreads();

  for (int j = threadIdx.x; j < NUM_CLASSES; j += 256)
    logits[(size_t)row * NUM_CLASSES + j] = hist[j];
}

// ---------------------------------------------------------------------------
extern "C" void kernel_launch(void* const* d_in, const int* in_sizes, int n_in,
                              void* d_out, int out_size, void* d_ws, size_t ws_size,
                              hipStream_t stream) {
  const float* x      = (const float*)d_in[0];   // [B, 512]
  const float* keys   = (const float*)d_in[1];   // [M, 512]
  const int*   values = (const int*)  d_in[2];   // [M]
  float* logits = (float*)d_out;                 // [B, 1000]

  const int B = in_sizes[0] / D_DIM;             // 1024
  const int M = in_sizes[2];                     // 131072
  const int nChunks = M / CHUNK;                 // 128

  // Workspace: xn (1 MB) | kn (128 MB, L2-resident during GEMM) | cand arrays
  char* ws = (char*)d_ws;
  size_t off = 0;
  __bf16* xn = (__bf16*)(ws + off);
  off += (size_t)B * D_DIM * sizeof(__bf16);   off = (off + 255) & ~(size_t)255;
  __bf16* kn = (__bf16*)(ws + off);
  off += (size_t)M * D_DIM * sizeof(__bf16);   off = (off + 255) & ~(size_t)255;
  float* candScore = (float*)(ws + off);
  off += (size_t)B * nChunks * K_NB * sizeof(float); off = (off + 255) & ~(size_t)255;
  int* candIdx = (int*)(ws + off);

  rownorm_bf16<<<(B + 7) / 8, 256, 0, stream>>>(x,    xn, B);
  rownorm_bf16<<<(M + 7) / 8, 256, 0, stream>>>(keys, kn, M);

  dim3 g3(B / ROWS, nChunks);                    // rows fastest -> chunk hot in L2
  gemm_topk<<<g3, 256, LDS_BYTES, stream>>>(xn, kn, values,
                                            candScore, candIdx, nChunks);

  merge_vote<<<B, 256, 0, stream>>>(candScore, candIdx, values, logits, nChunks);
}
